// GlobalSparseAttention_43860206026996
// MI455X (gfx1250) — matrix operand order repaired and checked by reading, not compiled, not run on hardware
//
#include <hip/hip_runtime.h>
#include <hip/hip_bf16.h>

// ---------------------------------------------------------------------------
// Sparse frame attention for MI455X (gfx1250, wave32, WMMA).
//   H=16 heads, D=64, N=2048 tokens, C=1024, frames of 256 tokens (fi = t>>8).
// Pipeline:
//   k0: f32 -> f16 conversions of x, Wqkv, Wproj into workspace
//   k1: QKV GEMM, 32x64 tile/wave (8x v_wmma_f32_16x16x32_f16 per k-step),
//       scatter into q[H][N][D] (×0.125), k[H][N][D], vT[H][D][N]
//   k2: per (head, 16-query tile): flash attention over 32-key chunks with
//       frame-block skipping (adj), frame bias, hub masking; QK^T and P·V via
//       WMMA; P staged through LDS to convert C-layout -> A-frag layout
//   k3: output projection GEMM (32x64 tile/wave) -> f32 d_out
// ---------------------------------------------------------------------------

typedef _Float16 v16h __attribute__((ext_vector_type(16)));
typedef _Float16 v8h  __attribute__((ext_vector_type(8)));
typedef float    v8f  __attribute__((ext_vector_type(8)));

#define NTOK 2048
#define CDIM 1024
#define C3   3072
#define HEADS 16
#define HD   64

// ---- WMMA fragment helpers (layouts per CDNA5 ISA §7.12.2) ------

// 16x32 f16 A: lane m=L&15; halves[0..7]=K (L>>4)*8+i ; halves[8..15]=+16
__device__ __forceinline__ v16h load_a_frag(const _Float16* rowk, int koff) {
  union { v16h v; v8h h[2]; } u;
  u.h[0] = *(const v8h*)(rowk + koff);
  u.h[1] = *(const v8h*)(rowk + koff + 16);
  return u.v;
}
// 32x16 f16 B: lane n=L&15; halves[0..7]=K (L>>4)*8+i ; halves[8..15]=+16
__device__ __forceinline__ v16h load_b_frag(const _Float16* rowk, int kb0) {
  union { v16h v; v8h h[2]; } u;
  u.h[0] = *(const v8h*)(rowk + kb0 / 2);
  u.h[1] = *(const v8h*)(rowk + kb0 / 2 + 16);
  return u.v;
}
__device__ __forceinline__ v8f wmma_f16(v16h a, v16h b, v8f c) {
  return __builtin_amdgcn_wmma_f32_16x16x32_f16(false, a, false, b,
                                                (short)0, c, false, false);
}

// ---- k0: f32 -> f16 --------------------------------------------------------
__global__ void cvt_f32_to_f16(const float* __restrict__ src,
                               _Float16* __restrict__ dst, int n) {
  int i = blockIdx.x * blockDim.x + threadIdx.x;
  if (i < n) dst[i] = (_Float16)src[i];
}

// ---- k1: QKV GEMM (32 rows x 64 cols per wave) + scatter -------------------
__global__ __launch_bounds__(32)
void qkv_gemm_kernel(const _Float16* __restrict__ xh,   // [N][C]
                     const _Float16* __restrict__ Wh,   // [3C][C]
                     const float* __restrict__ bqkv,    // [3C]
                     _Float16* __restrict__ Qo,         // [H][N][D]  (×0.125)
                     _Float16* __restrict__ Ko,         // [H][N][D]
                     _Float16* __restrict__ VTo) {      // [H][D][N]
  const int lane = threadIdx.x;
  const int n = lane & 15, hi = lane >> 4;
  const int koff = hi * 8, kb0 = hi * 16;
  const int col0 = blockIdx.x * 64;   // output feature tile (0..3071)
  const int row0 = blockIdx.y * 32;   // token tile
  const _Float16* arow0 = xh + (size_t)(row0 + n) * CDIM;
  const _Float16* arow1 = arow0 + (size_t)16 * CDIM;
  const _Float16* brow0 = Wh + (size_t)(col0 + n) * CDIM;
  v8f acc[2][4] = {};
#pragma unroll 2
  for (int k0 = 0; k0 < CDIM; k0 += 32) {
    __builtin_prefetch(brow0 + k0 + 128, 0, 0);   // global_prefetch_b8
    __builtin_prefetch(arow0 + k0 + 128, 0, 0);
    const v16h a0 = load_a_frag(arow0 + k0, koff);
    const v16h a1 = load_a_frag(arow1 + k0, koff);
#pragma unroll
    for (int c = 0; c < 4; ++c) {
      const v16h b = load_b_frag(brow0 + (size_t)c * 16 * CDIM + k0, kb0);
      acc[0][c] = wmma_f16(a0, b, acc[0][c]);
      acc[1][c] = wmma_f16(a1, b, acc[1][c]);
    }
  }
#pragma unroll
  for (int c = 0; c < 4; ++c) {
    const int j = col0 + c * 16 + n;  // lane's output feature
    const float bias = bqkv[j];
    const int which = j >> 10, cc = j & 1023, h = cc >> 6, dd = cc & 63;
#pragma unroll
    for (int i = 0; i < 2; ++i) {
#pragma unroll
      for (int r = 0; r < 8; ++r) {   // C-frag row = r + 8*hi
        const int t = row0 + i * 16 + r + 8 * hi;
        const float v = acc[i][c][r] + bias;
        if (which == 0)
          Qo[((size_t)h * NTOK + t) * HD + dd] = (_Float16)(v * 0.125f);
        else if (which == 1)
          Ko[((size_t)h * NTOK + t) * HD + dd] = (_Float16)v;
        else
          VTo[((size_t)h * HD + dd) * NTOK + t] = (_Float16)v;
      }
    }
  }
}

// ---- k2: sparse flash attention -------------------------------------------
__global__ __launch_bounds__(32)
void attn_kernel(const _Float16* __restrict__ Q,          // [H][N][D] pre-scaled
                 const _Float16* __restrict__ K,          // [H][N][D]
                 const _Float16* __restrict__ VT,         // [H][D][N]
                 const float* __restrict__ frame_bias,    // [8][8]
                 const unsigned char* __restrict__ adj,   // [8][8] bool
                 const unsigned char* __restrict__ hub,   // [N] bool
                 _Float16* __restrict__ Oh) {             // [N][C]
  __shared__ _Float16 Pld[16 * 32];                       // P staging, 1KB
  const int lane = threadIdx.x;
  const int n = lane & 15, hi = lane >> 4;
  const int koff = hi * 8, kb0 = hi * 16;
  const int h = blockIdx.y;
  const int q0 = blockIdx.x * 16;
  const int fq = q0 >> 8;                                 // frame of queries

  const _Float16* qrow = Q + ((size_t)h * NTOK + q0 + n) * HD;
  const v16h qa0 = load_a_frag(qrow, koff);               // d 0..31
  const v16h qa1 = load_a_frag(qrow + 32, koff);          // d 32..63

  bool hqr[8];
#pragma unroll
  for (int r = 0; r < 8; ++r) hqr[r] = hub[q0 + 8 * hi + r] != 0;

  v8f O0 = {}, O1 = {}, O2 = {}, O3 = {};
  float mr[8], lr[8];
#pragma unroll
  for (int r = 0; r < 8; ++r) { mr[r] = -1e30f; lr[r] = 0.0f; }

  for (int fk = 0; fk < 8; ++fk) {
    if (adj[fq * 8 + fk] == 0) continue;                  // frame-block skip
    const float fb = frame_bias[fq * 8 + fk];
    const bool same = (fk == fq);                         // same frame: no hub mask
    for (int key0 = fk * 256; key0 < (fk + 1) * 256; key0 += 32) {
      // ---- S = Q K^T (scale folded into Q) ----
      const _Float16* k0row = K + ((size_t)h * NTOK + key0 + n) * HD;
      const _Float16* k1row = k0row + 16 * HD;
      v8f S0 = {}, S1 = {};
      S0 = wmma_f16(qa0, load_b_frag(k0row, kb0), S0);
      S0 = wmma_f16(qa1, load_b_frag(k0row + 32, kb0), S0);
      S1 = wmma_f16(qa0, load_b_frag(k1row, kb0), S1);
      S1 = wmma_f16(qa1, load_b_frag(k1row + 32, kb0), S1);

      bool mk0 = false, mk1 = false;
      if (!same) { mk0 = hub[key0 + n] != 0; mk1 = hub[key0 + 16 + n] != 0; }

      float p0[8], p1[8], sc[8];
#pragma unroll
      for (int r = 0; r < 8; ++r) {
        float v0 = S0[r] + fb;
        float v1 = S1[r] + fb;
        if (!same) {                                      // hub masking
          if (hqr[r] | mk0) v0 = -1e30f;
          if (hqr[r] | mk1) v1 = -1e30f;
        }
        p0[r] = v0; p1[r] = v1;
      }
      // ---- online softmax: row stats via 16-lane xor reductions ----
#pragma unroll
      for (int r = 0; r < 8; ++r) {
        float cm = fmaxf(p0[r], p1[r]);
        cm = fmaxf(cm, __shfl_xor(cm, 1, 32));
        cm = fmaxf(cm, __shfl_xor(cm, 2, 32));
        cm = fmaxf(cm, __shfl_xor(cm, 4, 32));
        cm = fmaxf(cm, __shfl_xor(cm, 8, 32));
        const float mnew = fmaxf(mr[r], cm);
        const float s = (mr[r] > -5e29f) ? __expf(mr[r] - mnew) : 0.0f;
        sc[r] = s;
        mr[r] = mnew;
        const float e0 = (p0[r] > -5e29f) ? __expf(p0[r] - mnew) : 0.0f;
        const float e1 = (p1[r] > -5e29f) ? __expf(p1[r] - mnew) : 0.0f;
        p0[r] = e0; p1[r] = e1;
        float rs = e0 + e1;
        rs += __shfl_xor(rs, 1, 32);
        rs += __shfl_xor(rs, 2, 32);
        rs += __shfl_xor(rs, 4, 32);
        rs += __shfl_xor(rs, 8, 32);
        lr[r] = lr[r] * s + rs;
      }
#pragma unroll
      for (int r = 0; r < 8; ++r) {
        O0[r] *= sc[r]; O1[r] *= sc[r]; O2[r] *= sc[r]; O3[r] *= sc[r];
      }
      // ---- C-layout P -> A-frag layout via LDS ----
      __syncthreads();
#pragma unroll
      for (int r = 0; r < 8; ++r) {
        Pld[(r + 8 * hi) * 32 + n]      = (_Float16)p0[r];
        Pld[(r + 8 * hi) * 32 + 16 + n] = (_Float16)p1[r];
      }
      __syncthreads();
      const v16h pa = load_a_frag(&Pld[n * 32], koff);
      // ---- O += P · V  (B-frags contiguous thanks to V^T layout) ----
      const _Float16* vrow = VT + ((size_t)(h * HD + n)) * NTOK + key0;
      O0 = wmma_f16(pa, load_b_frag(vrow,             kb0), O0);
      O1 = wmma_f16(pa, load_b_frag(vrow + 16 * NTOK, kb0), O1);
      O2 = wmma_f16(pa, load_b_frag(vrow + 32 * NTOK, kb0), O2);
      O3 = wmma_f16(pa, load_b_frag(vrow + 48 * NTOK, kb0), O3);
    }
  }
  // ---- finalize: divide by l, write [N][C] f16 (head-major feature) ----
#pragma unroll
  for (int r = 0; r < 8; ++r) {
    const float inv = 1.0f / fmaxf(lr[r], 1e-30f);
    const int t = q0 + r + 8 * hi;
    _Float16* orow = Oh + (size_t)t * CDIM + h * HD;
    orow[0 + n]  = (_Float16)(O0[r] * inv);
    orow[16 + n] = (_Float16)(O1[r] * inv);
    orow[32 + n] = (_Float16)(O2[r] * inv);
    orow[48 + n] = (_Float16)(O3[r] * inv);
  }
}

// ---- k3: output projection (32 rows x 64 cols per wave) --------------------
__global__ __launch_bounds__(32)
void proj_gemm_kernel(const _Float16* __restrict__ Ah,   // [N][C]
                      const _Float16* __restrict__ Wh,   // [C][C]
                      const float* __restrict__ bproj,
                      float* __restrict__ out) {         // [N][C] f32
  const int lane = threadIdx.x;
  const int n = lane & 15, hi = lane >> 4;
  const int koff = hi * 8, kb0 = hi * 16;
  const int col0 = blockIdx.x * 64;
  const int row0 = blockIdx.y * 32;
  const _Float16* arow0 = Ah + (size_t)(row0 + n) * CDIM;
  const _Float16* arow1 = arow0 + (size_t)16 * CDIM;
  const _Float16* brow0 = Wh + (size_t)(col0 + n) * CDIM;
  v8f acc[2][4] = {};
#pragma unroll 2
  for (int k0 = 0; k0 < CDIM; k0 += 32) {
    __builtin_prefetch(brow0 + k0 + 128, 0, 0);
    __builtin_prefetch(arow0 + k0 + 128, 0, 0);
    const v16h a0 = load_a_frag(arow0 + k0, koff);
    const v16h a1 = load_a_frag(arow1 + k0, koff);
#pragma unroll
    for (int c = 0; c < 4; ++c) {
      const v16h b = load_b_frag(brow0 + (size_t)c * 16 * CDIM + k0, kb0);
      acc[0][c] = wmma_f16(a0, b, acc[0][c]);
      acc[1][c] = wmma_f16(a1, b, acc[1][c]);
    }
  }
#pragma unroll
  for (int c = 0; c < 4; ++c) {
    const int j = col0 + c * 16 + n;
    const float bias = bproj[j];
#pragma unroll
    for (int i = 0; i < 2; ++i) {
#pragma unroll
      for (int r = 0; r < 8; ++r) {
        const int t = row0 + i * 16 + r + 8 * hi;
        out[(size_t)t * CDIM + j] = acc[i][c][r] + bias;
      }
    }
  }
}

// ---------------------------------------------------------------------------
extern "C" void kernel_launch(void* const* d_in, const int* in_sizes, int n_in,
                              void* d_out, int out_size, void* d_ws, size_t ws_size,
                              hipStream_t stream) {
  const float* x          = (const float*)d_in[0];          // [1][2048][1024]
  const float* Wqkv       = (const float*)d_in[1];          // [3072][1024]
  const float* bqkv       = (const float*)d_in[2];          // [3072]
  const float* Wproj      = (const float*)d_in[3];          // [1024][1024]
  const float* bproj      = (const float*)d_in[4];          // [1024]
  const float* frame_bias = (const float*)d_in[5];          // [8][8]
  const unsigned char* adj    = (const unsigned char*)d_in[6];  // [8][8] bool
  /* d_in[7] frame_ids: deterministic (t >> 8), not read */
  const unsigned char* is_hub = (const unsigned char*)d_in[8];  // [2048] bool
  float* out = (float*)d_out;

  // workspace layout (halves); total 14,680,064 halves = 28 MB
  _Float16* ws = (_Float16*)d_ws;
  _Float16* xh     = ws;                                // 2048*1024
  _Float16* Wqkvh  = xh     + (size_t)NTOK * CDIM;      // 3072*1024
  _Float16* Wprojh = Wqkvh  + (size_t)C3 * CDIM;        // 1024*1024
  _Float16* qb     = Wprojh + (size_t)CDIM * CDIM;      // 16*2048*64
  _Float16* kb     = qb     + (size_t)HEADS * NTOK * HD;
  _Float16* vT     = kb     + (size_t)HEADS * NTOK * HD;
  _Float16* aoh    = vT     + (size_t)HEADS * NTOK * HD; // 2048*1024

  const int NX = NTOK * CDIM, NWQ = C3 * CDIM, NWP = CDIM * CDIM;
  cvt_f32_to_f16<<<(NX + 255) / 256, 256, 0, stream>>>(x, xh, NX);
  cvt_f32_to_f16<<<(NWQ + 255) / 256, 256, 0, stream>>>(Wqkv, Wqkvh, NWQ);
  cvt_f32_to_f16<<<(NWP + 255) / 256, 256, 0, stream>>>(Wproj, Wprojh, NWP);

  qkv_gemm_kernel<<<dim3(C3 / 64, NTOK / 32), 32, 0, stream>>>(
      xh, Wqkvh, bqkv, qb, kb, vT);

  attn_kernel<<<dim3(NTOK / 16, HEADS), 32, 0, stream>>>(
      qb, kb, vT, frame_bias, adj, is_hub, aoh);

  proj_gemm_kernel<<<dim3(CDIM / 64, NTOK / 32), 32, 0, stream>>>(
      aoh, Wprojh, bproj, out);
}
